// UNet3DModel_3212635537531
// MI455X (gfx1250) — matrix operand order repaired and checked by reading, not compiled, not run on hardware
//
#include <hip/hip_runtime.h>
#include <math.h>

// ============================================================================
// CDNA5 (gfx1250) dual-octree graph-UNet forward.
// Heavy math = graph_conv: structured gather (7 edges/node, node-owned, no
// atomics) -> f16-in/f32-acc WMMA GEMM (v_wmma_f32_16x16x32_f16).
// GEMM: 64x64 block tile, 4 waves x (32x32 wave tile) = 4 wmma/wave/chunk,
// vectorized global->LDS fill (fast path), b128 LDS fragment loads (B stored
// transposed in LDS so both A and B fragments are contiguous).
// ============================================================================

typedef _Float16 v16h __attribute__((ext_vector_type(16)));
typedef _Float16 v8h  __attribute__((ext_vector_type(8)));
typedef float    v8f  __attribute__((ext_vector_type(8)));

#define NEDGE 7
#define EPSGN 1e-5f

static inline int cdiv(int a, int b) { return (a + b - 1) / b; }

// ---------------------------------------------------------------------------
// Time embedding: t_emb[2,64] -> silu(t_emb@W1+b1)[2,256] -> silu(.@W2+b2)
// ---------------------------------------------------------------------------
__global__ __launch_bounds__(256) void time_embed_kernel(
    const float* __restrict__ ts, const float* __restrict__ w1,
    const float* __restrict__ b1, const float* __restrict__ w2,
    const float* __restrict__ b2, float* __restrict__ emb_silu)
{
  __shared__ float temb[2][64];
  __shared__ float e1[2][256];
  int t = threadIdx.x;
  if (t < 128) {
    int b = t >> 6, j = t & 63;
    float tv = ts[b];
    int i = (j < 32) ? j : (j - 32);
    float fr = expf(-logf(10000.f) * (float)i / 32.f);
    float a = tv * fr;
    temb[b][j] = (j < 32) ? cosf(a) : sinf(a);
  }
  __syncthreads();
  for (int b = 0; b < 2; ++b) {
    float acc = b1[t];
    for (int k = 0; k < 64; ++k) acc += temb[b][k] * w1[k * 256 + t];
    e1[b][t] = acc / (1.f + expf(-acc));   // silu
  }
  __syncthreads();
  for (int b = 0; b < 2; ++b) {
    float acc = b2[t];
    for (int k = 0; k < 256; ++k) acc += e1[b][k] * w2[k * 256 + t];
    emb_silu[b * 256 + t] = acc / (1.f + expf(-acc));  // silu(emb)
  }
}

// e[b, j] = silu_emb[b] @ W[256, Cout] + bias[j]
__global__ void emb_proj_kernel(const float* __restrict__ es,
                                const float* __restrict__ w,
                                const float* __restrict__ b,
                                float* __restrict__ e, int Cout)
{
  int i = blockIdx.x * blockDim.x + threadIdx.x;
  if (i >= 2 * Cout) return;
  int bb = i / Cout, j = i % Cout;
  float acc = b[j];
  for (int k = 0; k < 256; ++k) acc += es[bb * 256 + k] * w[k * Cout + j];
  e[bb * Cout + j] = acc;
}

__global__ void add_rows_kernel(float* __restrict__ h, const float* __restrict__ e,
                                const int* __restrict__ bid, int N, int C)
{
  int i = blockIdx.x * blockDim.x + threadIdx.x;
  if (i >= N * C) return;
  int n = i / C, c = i % C;
  h[i] += e[bid[n] * C + c];
}

__global__ void concat_kernel(const float* __restrict__ a, const float* __restrict__ b,
                              float* __restrict__ o, int N, int C1, int C2)
{
  int i = blockIdx.x * blockDim.x + threadIdx.x;
  int Ct = C1 + C2;
  if (i >= N * Ct) return;
  int n = i / Ct, c = i % Ct;
  o[i] = (c < C1) ? a[(size_t)n * C1 + c] : b[(size_t)n * C2 + (c - C1)];
}

// ---------------------------------------------------------------------------
// Gather/scatter for graph_conv. Each node owns edges [7n, 7n+7). One-hot of
// the source node type is generated on the fly. agg: [N, 7, C+ntw] row-major.
// ---------------------------------------------------------------------------
__global__ void gather_kernel(const float* __restrict__ x, const int* __restrict__ col,
                              const int* __restrict__ et, const int* __restrict__ nt,
                              float* __restrict__ agg, int N, int C, int ntw)
{
  int Cp = C + ntw;
  int i = blockIdx.x * blockDim.x + threadIdx.x;
  if (i >= N * Cp) return;
  int n = i / Cp, ch = i % Cp;
  int base = n * NEDGE;
  float acc[NEDGE] = {0.f, 0.f, 0.f, 0.f, 0.f, 0.f, 0.f};
  for (int e = 0; e < NEDGE; ++e) {
    int c = col[base + e];
    int t = et[base + e];
    float v = (ch < C) ? x[(size_t)c * C + ch]
                       : ((nt[c] == (ch - C)) ? 1.f : 0.f);
#pragma unroll
    for (int tt = 0; tt < NEDGE; ++tt) acc[tt] += (t == tt) ? v : 0.f;
  }
#pragma unroll
  for (int tt = 0; tt < NEDGE; ++tt)
    agg[((size_t)n * NEDGE + tt) * Cp + ch] = acc[tt];
}

// ---------------------------------------------------------------------------
// WMMA GEMM: out[M,Nc] = A[M,K] @ W[K,Nc] (+ addend[M,Nc]).
// Requires M % 64 == 0 (true for all call sites: 65536/8192/1024).
// 128 threads = 4 waves; block tile 64x64; wave tile 32x32 (4 accumulators).
// LDS: As[64][32] row-major, BsT[64][32] = B transposed (so fragment reads of
// both A and B are contiguous 8-half segments -> ds_load_b128).
// Fragment layouts per CDNA5 ISA (16-bit A 16x32; B 32x16; C/D f32).
// ---------------------------------------------------------------------------
#define APITCH 40  // halves; 80B row pitch, 16B-aligned
__global__ __launch_bounds__(128) void gemm_wmma_kernel(
    const float* __restrict__ A, const float* __restrict__ W,
    const float* __restrict__ addend, float* __restrict__ out,
    int M, int K, int Nc)
{
  __shared__ __align__(16) _Float16 As[64][APITCH];
  __shared__ __align__(16) _Float16 BsT[64][APITCH];
  int tid = threadIdx.x;
  int wave = tid >> 5;
  int lane = tid & 31;
  int wm = (wave >> 1) * 32;       // wave M offset within block tile
  int wn = (wave & 1) * 32;        // wave N offset
  int blockM = blockIdx.x * 64;
  int blockN = blockIdx.y * 64;
  int half16 = lane >> 4;
  int lrow = lane & 15;

  // fill assignments
  int ar = tid >> 1;               // A row 0..63
  int acb = (tid & 1) * 16;        // A col base {0,16}
  int bk = tid >> 2;               // B k-row 0..31
  int bcb = (tid & 3) * 16;        // B col base {0,16,32,48}

  v8f acc[2][2];
#pragma unroll
  for (int sm = 0; sm < 2; ++sm)
#pragma unroll
    for (int sn = 0; sn < 2; ++sn)
#pragma unroll
      for (int r = 0; r < 8; ++r) acc[sm][sn][r] = 0.f;

  const bool fullN = (blockN + 64 <= Nc);

  for (int k0 = 0; k0 < K; k0 += 32) {
    const bool fullK = (k0 + 32 <= K);
    // ---- A tile 64x32 ----
    {
      const float* Ar = A + (size_t)(blockM + ar) * K + k0 + acb;
      float tmp[16];
      if (fullK) {
#pragma unroll
        for (int j = 0; j < 16; ++j) tmp[j] = Ar[j];
      } else {
#pragma unroll
        for (int j = 0; j < 16; ++j) tmp[j] = (k0 + acb + j < K) ? Ar[j] : 0.f;
      }
      v8h p0, p1;
#pragma unroll
      for (int j = 0; j < 8; ++j) { p0[j] = (_Float16)tmp[j]; p1[j] = (_Float16)tmp[8 + j]; }
      *(v8h*)&As[ar][acb] = p0;
      *(v8h*)&As[ar][acb + 8] = p1;
    }
    // ---- B tile 32x64, stored transposed: BsT[n][k] ----
    {
      const float* Wr = W + (size_t)(k0 + bk) * Nc + blockN + bcb;
      if (fullK && fullN) {
#pragma unroll
        for (int j = 0; j < 16; ++j) BsT[bcb + j][bk] = (_Float16)Wr[j];
      } else {
        bool kok = (k0 + bk < K);
#pragma unroll
        for (int j = 0; j < 16; ++j) {
          float v = (kok && (blockN + bcb + j < Nc)) ? Wr[j] : 0.f;
          BsT[bcb + j][bk] = (_Float16)v;
        }
      }
    }
    __syncthreads();

    // ---- fragments (contiguous 8-half LDS reads) ----
    v16h af[2], bf[2];
#pragma unroll
    for (int sm = 0; sm < 2; ++sm) {
      int arow = wm + sm * 16 + lrow;
      v8h lo = *(const v8h*)&As[arow][half16 * 8];
      v8h hi = *(const v8h*)&As[arow][16 + half16 * 8];
#pragma unroll
      for (int j = 0; j < 8; ++j) { af[sm][j] = lo[j]; af[sm][8 + j] = hi[j]; }
    }
#pragma unroll
    for (int sn = 0; sn < 2; ++sn) {
      int bcol = wn + sn * 16 + lrow;
      v8h lo = *(const v8h*)&BsT[bcol][half16 * 8];
      v8h hi = *(const v8h*)&BsT[bcol][16 + half16 * 8];
#pragma unroll
      for (int j = 0; j < 8; ++j) { bf[sn][j] = lo[j]; bf[sn][8 + j] = hi[j]; }
    }
#pragma unroll
    for (int sm = 0; sm < 2; ++sm)
#pragma unroll
      for (int sn = 0; sn < 2; ++sn)
        acc[sm][sn] = __builtin_amdgcn_wmma_f32_16x16x32_f16(
            false, af[sm], false, bf[sn], (short)0, acc[sm][sn], false, false);
    __syncthreads();
  }

  // ---- store (M always in bounds; guard N only) ----
#pragma unroll
  for (int sm = 0; sm < 2; ++sm) {
#pragma unroll
    for (int sn = 0; sn < 2; ++sn) {
      int colg = blockN + wn + sn * 16 + lrow;
      if (colg < Nc) {
#pragma unroll
        for (int r = 0; r < 8; ++r) {
          int rowg = blockM + wm + sm * 16 + r + half16 * 8;
          float v = acc[sm][sn][r];
          if (addend) v += addend[(size_t)rowg * Nc + colg];
          out[(size_t)rowg * Nc + colg] = v;
        }
      }
    }
  }
}

// ---------------------------------------------------------------------------
// DualOctree GroupNorm (B=2, GROUPS=32, contiguous bid). Two-pass.
// ---------------------------------------------------------------------------
__global__ __launch_bounds__(256) void gn_stats_kernel(
    const float* __restrict__ x, const int* __restrict__ bid,
    float* __restrict__ stats, int N, int C)
{
  __shared__ float s_sum[256], s_sq[256];
  int t = threadIdx.x;
  int nodeInBlk = t >> 5;
  int g = t & 31;
  int n = blockIdx.x * 8 + nodeInBlk;
  int cg = C >> 5;
  float sum = 0.f, sq = 0.f;
  const float* xr = x + (size_t)n * C + g * cg;
  for (int i = 0; i < cg; ++i) { float v = xr[i]; sum += v; sq += v * v; }
  s_sum[t] = sum; s_sq[t] = sq;
  __syncthreads();
  if (t < 32) {
    float ts_ = 0.f, tq = 0.f;
    for (int j = 0; j < 8; ++j) { ts_ += s_sum[j * 32 + t]; tq += s_sq[j * 32 + t]; }
    int b = bid[blockIdx.x * 8];
    atomicAdd(&stats[(b * 32 + t) * 2 + 0], ts_);
    atomicAdd(&stats[(b * 32 + t) * 2 + 1], tq);
  }
}

__global__ void gn_apply_kernel(const float* __restrict__ x, const int* __restrict__ bid,
                                const float* __restrict__ stats,
                                const float* __restrict__ gamma,
                                const float* __restrict__ beta,
                                float* __restrict__ y, int N, int C,
                                float invcnt, int do_silu)
{
  int i = blockIdx.x * blockDim.x + threadIdx.x;
  if (i >= N * C) return;
  int n = i / C, c = i % C;
  int cg = C >> 5;
  int g = c / cg;
  int b = bid[n];
  float s0 = stats[(b * 32 + g) * 2 + 0];
  float s1 = stats[(b * 32 + g) * 2 + 1];
  float mean = s0 * invcnt;
  float var = s1 * invcnt - mean * mean;
  float inv = rsqrtf(var + EPSGN);
  float v = (x[i] - mean) * inv * gamma[c] + beta[c];
  if (do_silu) v = v / (1.f + expf(-v));
  y[i] = v;
}

// ===========================================================================
// Host orchestration
// ===========================================================================
// d_in flat layout: top-level setup_inputs() insertion order; 'params' pytree
// leaves in sorted-key DFS order:
//   0..3   down[0].conv, down[0].pool, down[1].conv, down[1].pool
//   4,5,6  end_b, end_g, in_conv
//   7..56  in_res[0..5] (b1,b2,conv1,conv2,emb_b,emb_w,g1,g2[,skip])
//          bases: 7,15,23(+skip),32,40(+skip),49
//   57     out_conv
//   58..138 out_res[0..8], 9 leaves each (all have skip), base = 58+9*i
//   139..142 time_b1, time_b2, time_w1, time_w2
//   143..146 up[0].conv, up[0].unpool, up[1].conv, up[1].unpool
//   147 x, 148 timesteps
//   149..153 row5,col5,etype5,ntype5,bid5 ; 154..158 d4 ; 159..163 d3

extern "C" void kernel_launch(void* const* d_in, const int* in_sizes, int n_in,
                              void* d_out, int out_size, void* d_ws, size_t ws_size,
                              hipStream_t stream)
{
  (void)in_sizes; (void)n_in; (void)out_size; (void)ws_size;

  auto f  = [&](int i) { return (const float*)d_in[i]; };
  auto ip = [&](int i) { return (const int*)d_in[i]; };

  int nodesArr[6] = {0}; int ntwArr[6] = {0};
  nodesArr[5] = 65536; nodesArr[4] = 8192; nodesArr[3] = 1024;
  ntwArr[5] = 4; ntwArr[4] = 3; ntwArr[3] = 2;
  const int* colP[6]; const int* etP[6]; const int* ntP[6]; const int* bidP[6];
  colP[5] = ip(150); etP[5] = ip(151); ntP[5] = ip(152); bidP[5] = ip(153);
  colP[4] = ip(155); etP[4] = ip(156); ntP[4] = ip(157); bidP[4] = ip(158);
  colP[3] = ip(160); etP[3] = ip(161); ntP[3] = ip(162); bidP[3] = ip(163);

  // ---- workspace carve-up (floats) ----
  float* WS = (float*)d_ws;
  size_t off = 0;
  auto alloc = [&](size_t n) { float* p = WS + off; off += n; return p; };
  float* EMB   = alloc(512);
  float* ETMP  = alloc(512);
  float* STATS = alloc(128);
  float* HS0 = alloc((size_t)65536 * 64);
  float* HS1 = alloc((size_t)65536 * 64);
  float* HS2 = alloc((size_t)65536 * 64);
  float* HS3 = alloc((size_t)8192 * 64);
  float* HS4 = alloc((size_t)8192 * 128);
  float* HS5 = alloc((size_t)8192 * 128);
  float* HS6 = alloc((size_t)1024 * 128);
  float* HS7 = alloc((size_t)1024 * 256);
  float* HS8 = alloc((size_t)1024 * 256);
  float* HA   = alloc((size_t)65536 * 128);
  float* CAT  = alloc((size_t)65536 * 192);
  float* HGN  = alloc((size_t)65536 * 192);
  float* H1   = alloc((size_t)65536 * 64);
  float* SKIP = alloc((size_t)65536 * 64);
  float* UNP  = alloc((size_t)8192 * 1024);
  float* AGG  = alloc((size_t)65536 * 1372);

  // ---- primitive launchers ----
  auto gemmF = [&](const float* A, const float* Wm, const float* add, float* out,
                   int M, int K, int Nc) {
    dim3 grid(M / 64, cdiv(Nc, 64));
    gemm_wmma_kernel<<<grid, 128, 0, stream>>>(A, Wm, add, out, M, K, Nc);
  };
  auto gatherF = [&](const float* x, int d, int C) {
    int N = nodesArr[d], ntw = ntwArr[d];
    int total = N * (C + ntw);
    gather_kernel<<<cdiv(total, 256), 256, 0, stream>>>(
        x, colP[d], etP[d], ntP[d], AGG, N, C, ntw);
  };
  auto convF = [&](const float* x, int d, int C, int Cout, const float* Wm,
                   const float* add, float* out) {
    gatherF(x, d, C);
    gemmF(AGG, Wm, add, out, nodesArr[d], NEDGE * (C + ntwArr[d]), Cout);
  };
  auto gnF = [&](const float* x, int d, int C, const float* g_, const float* b_,
                 float* y, int silu) {
    int N = nodesArr[d];
    hipMemsetAsync(STATS, 0, 128 * sizeof(float), stream);
    gn_stats_kernel<<<N / 8, 256, 0, stream>>>(x, bidP[d], STATS, N, C);
    float invcnt = 1.0f / ((float)(N / 2) * (float)(C / 32));
    gn_apply_kernel<<<cdiv(N * C, 256), 256, 0, stream>>>(
        x, bidP[d], STATS, g_, b_, y, N, C, invcnt, silu);
  };
  auto resF = [&](const float* x, int d, int cin, int cout, int base,
                  bool hasSkip, float* out) {
    int N = nodesArr[d];
    gnF(x, d, cin, f(base + 6), f(base + 0), HGN, 1);            // g1,b1 + silu
    convF(HGN, d, cin, cout, f(base + 2), nullptr, H1);          // conv1
    emb_proj_kernel<<<cdiv(2 * cout, 64), 64, 0, stream>>>(
        EMB, f(base + 5), f(base + 4), ETMP, cout);              // emb_w, emb_b
    add_rows_kernel<<<cdiv(N * cout, 256), 256, 0, stream>>>(H1, ETMP, bidP[d], N, cout);
    gnF(H1, d, cout, f(base + 7), f(base + 1), HGN, 1);          // g2,b2 + silu
    const float* add = x;
    if (hasSkip) { gemmF(x, f(base + 8), nullptr, SKIP, N, cin, cout); add = SKIP; }
    convF(HGN, d, cout, cout, f(base + 3), add, out);            // conv2 + skip
  };
  auto catF = [&](const float* a, int C1, const float* b, int C2, int N, float* o) {
    int total = N * (C1 + C2);
    concat_kernel<<<cdiv(total, 256), 256, 0, stream>>>(a, b, o, N, C1, C2);
  };

  // ---- time embedding ----
  time_embed_kernel<<<1, 256, 0, stream>>>(f(148), f(141), f(139), f(142), f(140), EMB);

  // ---- encoder ----
  convF(f(147), 5, 8, 64, f(6), nullptr, HS0);                   // in_conv
  resF(HS0, 5, 64, 64, 7, false, HS1);                           // in_res[0]
  resF(HS1, 5, 64, 64, 15, false, HS2);                          // in_res[1]
  gemmF(HS2, f(1), nullptr, UNP, 8192, 512, 64);                 // down0.pool
  convF(UNP, 4, 64, 64, f(0), nullptr, HS3);                     // down0.conv
  resF(HS3, 4, 64, 128, 23, true, HS4);                          // in_res[2]
  resF(HS4, 4, 128, 128, 32, false, HS5);                        // in_res[3]
  gemmF(HS5, f(3), nullptr, UNP, 1024, 1024, 128);               // down1.pool
  convF(UNP, 3, 128, 128, f(2), nullptr, HS6);                   // down1.conv
  resF(HS6, 3, 128, 256, 40, true, HS7);                         // in_res[4]
  resF(HS7, 3, 256, 256, 49, false, HS8);                        // in_res[5]

  // ---- decoder ----
  const int OR0 = 58;
  catF(HS8, 256, HS8, 256, 1024, CAT);
  resF(CAT, 3, 512, 256, OR0 + 0, true, HA);                     // out_res[0]
  catF(HA, 256, HS7, 256, 1024, CAT);
  resF(CAT, 3, 512, 256, OR0 + 9, true, HA);                     // out_res[1]
  catF(HA, 256, HS6, 128, 1024, CAT);
  resF(CAT, 3, 384, 256, OR0 + 18, true, HA);                    // out_res[2]
  gemmF(HA, f(144), nullptr, UNP, 1024, 256, 2048);              // up0.unpool
  convF(UNP, 4, 256, 256, f(143), nullptr, HA);                  // up0.conv -> [8192,256]
  catF(HA, 256, HS5, 128, 8192, CAT);
  resF(CAT, 4, 384, 128, OR0 + 27, true, HA);                    // out_res[3]
  catF(HA, 128, HS4, 128, 8192, CAT);
  resF(CAT, 4, 256, 128, OR0 + 36, true, HA);                    // out_res[4]
  catF(HA, 128, HS3, 64, 8192, CAT);
  resF(CAT, 4, 192, 128, OR0 + 45, true, HA);                    // out_res[5]
  gemmF(HA, f(146), nullptr, UNP, 8192, 128, 1024);              // up1.unpool
  convF(UNP, 5, 128, 128, f(145), nullptr, HA);                  // up1.conv -> [65536,128]
  catF(HA, 128, HS2, 64, 65536, CAT);
  resF(CAT, 5, 192, 64, OR0 + 54, true, HA);                     // out_res[6]
  catF(HA, 64, HS1, 64, 65536, CAT);
  resF(CAT, 5, 128, 64, OR0 + 63, true, HA);                     // out_res[7]
  catF(HA, 64, HS0, 64, 65536, CAT);
  resF(CAT, 5, 128, 64, OR0 + 72, true, HA);                     // out_res[8]

  // ---- head ----
  gnF(HA, 5, 64, f(5), f(4), HGN, 1);                            // end_g, end_b + silu
  convF(HGN, 5, 64, 8, f(57), nullptr, (float*)d_out);           // out_conv -> [65536, 8]
}
